// SwinTransformerBlock_6743098655488
// MI455X (gfx1250) — compile-verified, matrix operand order checked by reading, not run on hardware
//
#include <hip/hip_runtime.h>
#include <hip/hip_bf16.h>

// ---------------------------------------------------------------------------
// Swin block, MI455X (gfx1250, wave32). Fully fused: one workgroup per 8x8
// window does norm1 -> QKV -> attention -> proj(+residual) -> norm2 ->
// fc1+GELU -> fc2(+residual). All intermediates live in a phase-overlapped
// 124 KB LDS arena (CDNA5 WGP LDS); only x (read) and out (write) hit HBM.
// ---------------------------------------------------------------------------
typedef __attribute__((ext_vector_type(16))) _Float16 v16h;
typedef __attribute__((ext_vector_type(8)))  float    v8f;

#define DEVINL __device__ __forceinline__

// 16x32 f16 WMMA A/B fragment (works for global or LDS pointers).
//   p  = row start (+kk), hk = (lane>>4)*8 per the 16-bit operand layout:
//   elems 0..7 <- K = hk..hk+7 ; elems 8..15 <- K = 16+hk..16+hk+7
DEVINL v16h ldfrag(const _Float16* p, int hk) {
    union { v16h v; float4 q[2]; } u;
    u.q[0] = *reinterpret_cast<const float4*>(p + hk);
    u.q[1] = *reinterpret_cast<const float4*>(p + hk + 16);
    return u.v;
}

DEVINL v8f wmma16(v16h a, v16h b, v8f c) {
    return __builtin_amdgcn_wmma_f32_16x16x32_f16(
        false, a, false, b, (short)0, c, false, false);
}

// token index for row t (0..63) of window `win`
DEVINL int tok_of(int win, int t) {
    int b = win >> 6, wloc = win & 63;
    int wh = wloc >> 3, ww = wloc & 7;
    int th = t >> 3, tw = t & 7;
    return b * 4096 + (wh * 8 + th) * 64 + (ww * 8 + tw);
}

// ---------------------------------------------------------------------------
// K0: W[K,N] f32 -> WT[N,K] f16 so WMMA B-operand columns are contiguous.
// ---------------------------------------------------------------------------
__global__ __launch_bounds__(256) void transpose_to_f16(
    const float* __restrict__ W, _Float16* __restrict__ WT, int K, int N) {
    int idx = blockIdx.x * 256 + threadIdx.x;
    if (idx < K * N) {
        int k = idx / N, n = idx % N;
        WT[(size_t)n * K + k] = (_Float16)W[idx];
    }
}

// ---------------------------------------------------------------------------
// K1: the fused Swin block. grid = 2048 windows, block = 256 threads (8 waves)
// ---------------------------------------------------------------------------
__global__ __launch_bounds__(256) void swin_block_fused(
    const float* __restrict__ x,
    const float* __restrict__ n1g, const float* __restrict__ n1b,
    const _Float16* __restrict__ qkvT, const float* __restrict__ qkv_b,
    const _Float16* __restrict__ projT, const float* __restrict__ proj_b,
    const float* __restrict__ n2g, const float* __restrict__ n2b,
    const _Float16* __restrict__ fc1T, const float* __restrict__ b1,
    const _Float16* __restrict__ fc2T, const float* __restrict__ b2,
    float* __restrict__ out) {

    // ---- phase-overlapped LDS arena (124 KB) ----
    __shared__ __align__(16) unsigned char smem[126976];
    _Float16* hb      = (_Float16*)(smem + 0);       // LN1 out   [64*192] f16
    _Float16* attnbuf = (_Float16*)(smem + 0);       // attn out  [64*192] f16
    _Float16* qkvt    = (_Float16*)(smem + 24576);   // qkv tile  [64*576] f16
    float*    x1ls    = (float*)   (smem + 24576);   // x1        [64*192] f32
    _Float16* hb2     = (_Float16*)(smem + 73728);   // LN2 out   [64*192] f16
    float*    S       = (float*)   (smem + 98304);   // scores    [64*64]  f32
    _Float16* hiddenC = (_Float16*)(smem + 98304);   // mlp chunk [64*192] f16
    _Float16* P       = (_Float16*)(smem + 114688);  // softmax   [64*64]  f16
    _Float16* VT      = (_Float16*)(smem + 122880);  // V^T       [32*64]  f16

    int win  = blockIdx.x;
    int tid  = threadIdx.x, wv = tid >> 5, lane = tid & 31;
    int ln   = lane & 15, hk = (lane >> 4) << 3;
    int mi   = wv & 3;                 // M-subtile for every GEMM phase
    int grp6 = (wv >> 2) * 6;          // N-tile group base (6 tiles/wave)

    // ============ LN1 (+ window gather): each wave does 8 rows ============
    for (int rr = 0; rr < 8; ++rr) {
        int rl = wv * 8 + rr;
        const float* row = x + (size_t)tok_of(win, rl) * 192;
        float vals[6], s = 0.f, ss = 0.f;
        for (int i = 0; i < 6; ++i) {
            float t = row[lane + 32 * i];
            vals[i] = t; s += t; ss += t * t;
        }
        for (int m = 16; m > 0; m >>= 1) {
            s  += __shfl_xor(s,  m, 32);
            ss += __shfl_xor(ss, m, 32);
        }
        float mean = s * (1.f / 192.f);
        float rs   = rsqrtf(ss * (1.f / 192.f) - mean * mean + 1e-5f);
        for (int i = 0; i < 6; ++i) {
            int cc = lane + 32 * i;
            hb[rl * 192 + cc] = (_Float16)((vals[i] - mean) * rs * n1g[cc] + n1b[cc]);
        }
    }
    __syncthreads();

    // ============ QKV GEMM: [64,192] @ [192,576] -> qkvt (LDS) ============
    {
        const _Float16* arow = hb + (mi * 16 + ln) * 192;
        int niStart = (wv >> 2) * 18;              // 36 N-tiles, 18/wave
        for (int ni = niStart; ni < niStart + 18; ++ni) {
            v8f c = {};
            const _Float16* brow = qkvT + (size_t)(ni * 16 + ln) * 192;
            for (int kk = 0; kk < 192; kk += 32)
                c = wmma16(ldfrag(arow + kk, hk), ldfrag(brow + kk, hk), c);
            int n = ni * 16 + ln;
            float bb = qkv_b[n];
            for (int v = 0; v < 8; ++v)
                qkvt[(mi * 16 + v + hk) * 576 + n] = (_Float16)(c[v] + bb);
        }
    }
    __syncthreads();

    // ============ attention: 6 heads, hd=32 (one WMMA K-step) ============
    const float scale = 0.17677669529663687f;      // 1/sqrt(32)
    for (int h = 0; h < 6; ++h) {
        if (h) __syncthreads();                    // protect S/P/VT reuse
        // stage V^T (so V columns become contiguous B-operand rows)
        for (int e = 0; e < 8; ++e) {
            int id = tid * 8 + e;                  // 0..2047
            int r = id >> 5, cch = id & 31;
            VT[cch * 64 + r] = qkvt[r * 576 + 384 + h * 32 + cch];
        }
        // S = Q K^T * scale : 16 tiles, 2 per wave
        int nb = (wv >> 2) * 2;
        v16h qa = ldfrag(qkvt + (mi * 16 + ln) * 576 + h * 32, hk);
        for (int j = 0; j < 2; ++j) {
            int nt = nb + j;
            v16h kb = ldfrag(qkvt + (nt * 16 + ln) * 576 + 192 + h * 32, hk);
            v8f c = {};
            c = wmma16(qa, kb, c);
            int n = nt * 16 + ln;
            for (int v = 0; v < 8; ++v)
                S[(mi * 16 + v + hk) * 64 + n] = c[v] * scale;
        }
        __syncthreads();
        // softmax: one thread per row
        if (tid < 64) {
            float mx = -1e30f;
            for (int j = 0; j < 64; ++j) mx = fmaxf(mx, S[tid * 64 + j]);
            float sum = 0.f;
            for (int j = 0; j < 64; ++j) {
                float e = __expf(S[tid * 64 + j] - mx);
                S[tid * 64 + j] = e; sum += e;
            }
            float inv = 1.f / sum;
            for (int j = 0; j < 64; ++j)
                P[tid * 64 + j] = (_Float16)(S[tid * 64 + j] * inv);
        }
        __syncthreads();
        // O = P V : 8 tiles, one per wave; K=64 -> 2 steps
        int nO = wv >> 2;
        v8f o = {};
        for (int kk = 0; kk < 64; kk += 32)
            o = wmma16(ldfrag(P  + (mi * 16 + ln) * 64 + kk, hk),
                       ldfrag(VT + (nO * 16 + ln) * 64 + kk, hk), o);
        int col = h * 32 + nO * 16 + ln;
        for (int v = 0; v < 8; ++v)
            attnbuf[(mi * 16 + v + hk) * 192 + col] = (_Float16)o[v];
    }
    __syncthreads();

    // ============ proj + bias + shortcut residual -> x1 (LDS, f32) ========
    {
        const _Float16* arow = attnbuf + (mi * 16 + ln) * 192;
        for (int ni = grp6; ni < grp6 + 6; ++ni) {     // 12 N-tiles, 6/wave
            v8f c = {};
            const _Float16* brow = projT + (size_t)(ni * 16 + ln) * 192;
            for (int kk = 0; kk < 192; kk += 32)
                c = wmma16(ldfrag(arow + kk, hk), ldfrag(brow + kk, hk), c);
            int col = ni * 16 + ln;
            float bb = proj_b[col];
            for (int v = 0; v < 8; ++v) {
                int t = mi * 16 + v + hk;
                x1ls[t * 192 + col] =
                    c[v] + bb + x[(size_t)tok_of(win, t) * 192 + col];
            }
        }
    }
    __syncthreads();

    // ============ LN2 -> hb2 (LDS f16) ============
    for (int rr = 0; rr < 8; ++rr) {
        int rl = wv * 8 + rr;
        float vals[6], s = 0.f, ss = 0.f;
        for (int i = 0; i < 6; ++i) {
            float t = x1ls[rl * 192 + lane + 32 * i];
            vals[i] = t; s += t; ss += t * t;
        }
        for (int m = 16; m > 0; m >>= 1) {
            s  += __shfl_xor(s,  m, 32);
            ss += __shfl_xor(ss, m, 32);
        }
        float mean = s * (1.f / 192.f);
        float rs   = rsqrtf(ss * (1.f / 192.f) - mean * mean + 1e-5f);
        for (int i = 0; i < 6; ++i) {
            int cc = lane + 32 * i;
            hb2[rl * 192 + cc] = (_Float16)((vals[i] - mean) * rs * n2g[cc] + n2b[cc]);
        }
    }
    __syncthreads();

    // ============ MLP: fc1+GELU / fc2, hidden chunked 4 x 192 cols ========
    v8f acc2[6] = {};                               // fc2 accumulators
    for (int ch = 0; ch < 4; ++ch) {
        int c0 = ch * 192;
        // fc1 chunk: [64,192] @ [192,192-chunk] -> hiddenC (with exact GELU)
        {
            const _Float16* arow = hb2 + (mi * 16 + ln) * 192;
            for (int ni = grp6; ni < grp6 + 6; ++ni) {
                v8f c = {};
                const _Float16* brow = fc1T + (size_t)(c0 + ni * 16 + ln) * 192;
                for (int kk = 0; kk < 192; kk += 32)
                    c = wmma16(ldfrag(arow + kk, hk), ldfrag(brow + kk, hk), c);
                int nloc = ni * 16 + ln;
                float bb = b1[c0 + nloc];
                for (int v = 0; v < 8; ++v) {
                    float hv = c[v] + bb;
                    float ge = 0.5f * hv * (1.f + erff(hv * 0.70710678118f));
                    hiddenC[(mi * 16 + v + hk) * 192 + nloc] = (_Float16)ge;
                }
            }
        }
        __syncthreads();
        // fc2 partial: accumulate K-chunk c0..c0+191
        {
            const _Float16* arow = hiddenC + (mi * 16 + ln) * 192;
#pragma unroll
            for (int j = 0; j < 6; ++j) {
                int ni = grp6 + j;
                const _Float16* brow = fc2T + (size_t)(ni * 16 + ln) * 768 + c0;
                for (int kk = 0; kk < 192; kk += 32)
                    acc2[j] = wmma16(ldfrag(arow + kk, hk),
                                     ldfrag(brow + kk, hk), acc2[j]);
            }
        }
        __syncthreads();                            // before hiddenC overwrite
    }

    // ============ final: out[token] = x1 + fc2 + b2 (un-window scatter) ====
#pragma unroll
    for (int j = 0; j < 6; ++j) {
        int col = (grp6 + j) * 16 + ln;
        float bb = b2[col];
        for (int v = 0; v < 8; ++v) {
            int t = mi * 16 + v + hk;
            out[(size_t)tok_of(win, t) * 192 + col] =
                acc2[j][v] + bb + x1ls[t * 192 + col];
        }
    }
}

// ---------------------------------------------------------------------------
// launch
// ---------------------------------------------------------------------------
extern "C" void kernel_launch(void* const* d_in, const int* in_sizes, int n_in,
                              void* d_out, int out_size, void* d_ws, size_t ws_size,
                              hipStream_t stream) {
    const float* x      = (const float*)d_in[0];
    const float* n1g    = (const float*)d_in[1];
    const float* n1b    = (const float*)d_in[2];
    const float* qkv_w  = (const float*)d_in[3];
    const float* qkv_b  = (const float*)d_in[4];
    const float* proj_w = (const float*)d_in[5];
    const float* proj_b = (const float*)d_in[6];
    const float* n2g    = (const float*)d_in[7];
    const float* n2b    = (const float*)d_in[8];
    const float* fc1_w  = (const float*)d_in[9];
    const float* fc1_b  = (const float*)d_in[10];
    const float* fc2_w  = (const float*)d_in[11];
    const float* fc2_b  = (const float*)d_in[12];
    float* out = (float*)d_out;

    // workspace: f16 transposed weights only (<1 MB)
    char* ws = (char*)d_ws;
    _Float16* qkvT  = (_Float16*)(ws + 0);        // [576][192]
    _Float16* projT = (_Float16*)(ws + 221184);   // [192][192]
    _Float16* fc1T  = (_Float16*)(ws + 294912);   // [768][192]
    _Float16* fc2T  = (_Float16*)(ws + 589824);   // [192][768]

    transpose_to_f16<<<(192 * 576 + 255) / 256, 256, 0, stream>>>(qkv_w,  qkvT, 192, 576);
    transpose_to_f16<<<(192 * 192 + 255) / 256, 256, 0, stream>>>(proj_w, projT, 192, 192);
    transpose_to_f16<<<(192 * 768 + 255) / 256, 256, 0, stream>>>(fc1_w,  fc1T, 192, 768);
    transpose_to_f16<<<(768 * 192 + 255) / 256, 256, 0, stream>>>(fc2_w,  fc2T, 768, 192);

    swin_block_fused<<<2048, 256, 0, stream>>>(
        x, n1g, n1b, qkvT, qkv_b, projT, proj_b,
        n2g, n2b, fc1T, fc1_b, fc2T, fc2_b, out);
}
